// GAT_1846835937321
// MI455X (gfx1250) — matrix operand order repaired and checked by reading, not compiled, not run on hardware
//
#include <hip/hip_runtime.h>
#include <stdint.h>

#define HH   3
#define FIN  128
#define C1   96    // H*F1
#define F2D  32
#define C2   96    // H*F2 (pre-mean layer-2 width)
#define NT   6     // column tiles (96/16) for both GEMMs
#define CDIV(a,b) (((a)+(b)-1)/(b))

typedef _Float16 h8   __attribute__((ext_vector_type(8)));
typedef _Float16 v16h __attribute__((ext_vector_type(16)));
typedef float    v8f  __attribute__((ext_vector_type(8)));

// ---------------- elementwise utils ----------------
__global__ void k_fill_u32(uint32_t* __restrict__ p, uint32_t v, int n) {
  int i = blockIdx.x * blockDim.x + threadIdx.x;
  if (i < n) p[i] = v;
}

__global__ void k_f32_to_f16(const float* __restrict__ in, _Float16* __restrict__ out, int n) {
  int i = blockIdx.x * blockDim.x + threadIdx.x;
  if (i < n) out[i] = (_Float16)in[i];
}

// ---------------- pack W [K x Ncol] (f32) into WMMA B-fragment order (f16) ----------------
// Pack index p = ((tn*KSTEPS + kk)*32 + lane)*16 + i
// holds W[(kk*32 + 8*(lane>>4) + (i&7) + 16*(i>>3)) * Ncol + tn*16 + (lane&15)]
__global__ void k_pack_B(const float* __restrict__ W, _Float16* __restrict__ Bp,
                         int K, int Ncol) {
  int idx = blockIdx.x * blockDim.x + threadIdx.x;
  if (idx >= K * Ncol) return;
  int i   = idx & 15;
  int l   = (idx >> 4) & 31;
  int t   = idx >> 9;                 // tn*KSTEPS + kk
  int ksteps = K >> 5;
  int tn  = t / ksteps, kk = t - tn * ksteps;
  int l15 = l & 15, g = l >> 4;
  int Kidx = kk * 32 + 8 * g + (i & 7) + 16 * (i >> 3);
  int col  = tn * 16 + l15;
  Bp[idx] = (_Float16)W[(size_t)Kidx * Ncol + col];
}

// ---------------- WMMA GEMM: C[M x 96] = A[M x K](f16) * B(packed f16), f32 accum ----------
// One wave computes a full 16-row x 96-col stripe: A fragment reused across 6 column tiles.
template<int KSTEPS>
__global__ void k_gemm_wmma(const _Float16* __restrict__ A,
                            const _Float16* __restrict__ Bp,
                            float* __restrict__ C, int M) {
  const int wavesPerBlock = blockDim.x >> 5;
  const int wave = blockIdx.x * wavesPerBlock + (threadIdx.x >> 5);
  const int lane = threadIdx.x & 31;
  const int rowTiles = CDIV(M, 16);
  if (wave >= rowTiles) return;        // wave-uniform; EXEC stays all-ones for WMMA

  const int row0 = wave << 4;
  const int l15 = lane & 15, g = lane >> 4;
  const int K = KSTEPS * 32;

  int ar = row0 + l15; if (ar > M - 1) ar = M - 1;
  const _Float16* Arow = A + (size_t)ar * K;

  v8f acc[NT];
  #pragma unroll
  for (int t = 0; t < NT; ++t) acc[t] = (v8f){};

  #pragma unroll
  for (int kk = 0; kk < KSTEPS; ++kk) {
    // A fragment: lane group g holds K = {32kk+8g .. +7} and {32kk+16+8g .. +7}
    h8 alo = *(const h8*)(Arow + kk * 32 + 8 * g);
    h8 ahi = *(const h8*)(Arow + kk * 32 + 16 + 8 * g);
    v16h a = __builtin_shufflevector(alo, ahi, 0, 1, 2, 3, 4, 5, 6, 7,
                                     8, 9, 10, 11, 12, 13, 14, 15);
    #pragma unroll
    for (int tn = 0; tn < NT; ++tn) {
      v16h b = *(const v16h*)(Bp + ((size_t)(tn * KSTEPS + kk) * 32 + lane) * 16);
      acc[tn] = __builtin_amdgcn_wmma_f32_16x16x32_f16(false, a, false, b,
                                                       (short)0, acc[tn], false, false);
    }
  }

  // C/D layout: VGPR i -> row = i + 8*g, col = l15
  #pragma unroll
  for (int tn = 0; tn < NT; ++tn) {
    #pragma unroll
    for (int i = 0; i < 8; ++i) {
      int rr = row0 + i + 8 * g;
      if (rr < M) C[(size_t)rr * 96 + tn * 16 + l15] = acc[tn][i];
    }
  }
}

// ---------------- attention score reduction: a_s/a_d [N,H] ----------------
__global__ void k_attn_scores(const float* __restrict__ hfeat,   // [N, H*F]
                              const float* __restrict__ att_src, // [H, F]
                              const float* __restrict__ att_dst, // [H, F]
                              float* __restrict__ a_s, float* __restrict__ a_d,
                              int N, int F) {
  int idx = blockIdx.x * blockDim.x + threadIdx.x;        // over N*H
  if (idx >= N * HH) return;
  int n = idx / HH, h = idx - n * HH;
  const float* hp  = hfeat + (size_t)n * (HH * F) + h * F;
  const float* sp  = att_src + h * F;
  const float* dp  = att_dst + h * F;
  float ss = 0.f, sd = 0.f;
  #pragma unroll 8
  for (int f = 0; f < F; ++f) { float v = hp[f]; ss += v * sp[f]; sd += v * dp[f]; }
  a_s[idx] = ss; a_d[idx] = sd;
}

// ---------------- edge passes ----------------
__device__ inline void atomicMaxF(float* addr, float v) {
  if (v >= 0.f) atomicMax((int*)addr, __float_as_int(v));
  else          atomicMin((unsigned int*)addr, __float_as_uint(v));
}

__device__ inline void edge_decode(int e, const int* src, const int* dst, int E,
                                   int& s, int& d) {
  if (e < E) { s = src[e]; d = dst[e]; } else { s = e - E; d = e - E; }  // self-loops
}

// pass A: logits (LeakyReLU 0.2) -> elog; running segment max into emax
__global__ void k_edge_max(const int* __restrict__ src, const int* __restrict__ dst,
                           int E, int Etot,
                           const float* __restrict__ a_s, const float* __restrict__ a_d,
                           float* __restrict__ elog, float* __restrict__ emax) {
  int e = blockIdx.x * blockDim.x + threadIdx.x;
  if (e >= Etot) return;
  int s, d; edge_decode(e, src, dst, E, s, d);
  #pragma unroll
  for (int h = 0; h < HH; ++h) {
    float v = a_s[s * HH + h] + a_d[d * HH + h];
    v = v > 0.f ? v : 0.2f * v;
    elog[(size_t)e * HH + h] = v;
    atomicMaxF(&emax[d * HH + h], v);
  }
}

// pass B: ex = exp(e - max), accumulate denom; elog overwritten with ex
__global__ void k_edge_expsum(const int* __restrict__ src, const int* __restrict__ dst,
                              int E, int Etot,
                              const float* __restrict__ emax,
                              float* __restrict__ elog, float* __restrict__ denom) {
  int e = blockIdx.x * blockDim.x + threadIdx.x;
  if (e >= Etot) return;
  int s, d; edge_decode(e, src, dst, E, s, d);
  #pragma unroll
  for (int h = 0; h < HH; ++h) {
    float ex = expf(elog[(size_t)e * HH + h] - emax[d * HH + h]);
    elog[(size_t)e * HH + h] = ex;
    atomicAdd(&denom[d * HH + h], ex);
  }
}

// pass C: alpha-weighted scatter of source features (per edge-head thread, float4 gathers)
__global__ void k_edge_aggregate(const int* __restrict__ src, const int* __restrict__ dst,
                                 int E, int Etot,
                                 const float* __restrict__ elog, const float* __restrict__ denom,
                                 const float* __restrict__ feat,   // [N, H*32]
                                 float* __restrict__ acc) {        // [N, H*32]
  int idx = blockIdx.x * blockDim.x + threadIdx.x;        // over Etot*H
  if (idx >= Etot * HH) return;
  int e = idx / HH, h = idx - e * HH;
  int s, d; edge_decode(e, src, dst, E, s, d);
  float alpha = elog[(size_t)e * HH + h] / denom[d * HH + h];
  const float4* hp = (const float4*)(feat + (size_t)s * C1 + h * 32);
  float* op        = acc + (size_t)d * C1 + h * 32;
  #pragma unroll
  for (int q = 0; q < 8; ++q) {
    float4 v = hp[q];
    atomicAdd(op + 4 * q + 0, alpha * v.x);
    atomicAdd(op + 4 * q + 1, alpha * v.y);
    atomicAdd(op + 4 * q + 2, alpha * v.z);
    atomicAdd(op + 4 * q + 3, alpha * v.w);
  }
}

// ---------------- finalize ----------------
// layer 1: relu(acc + b1) -> f16 for next GEMM (concat layout already correct)
__global__ void k_finalize_concat(const float* __restrict__ acc, const float* __restrict__ bias,
                                  _Float16* __restrict__ out, int N) {
  int idx = blockIdx.x * blockDim.x + threadIdx.x;        // over N*C1
  if (idx >= N * C1) return;
  int c = idx % C1;
  float v = acc[idx] + bias[c];
  out[idx] = (_Float16)(v > 0.f ? v : 0.f);
}

// layer 2: mean over heads, + b2, relu -> f32 [N, F2]
__global__ void k_finalize_mean(const float* __restrict__ acc, const float* __restrict__ bias,
                                float* __restrict__ out, int N) {
  int idx = blockIdx.x * blockDim.x + threadIdx.x;        // over N*F2
  if (idx >= N * F2D) return;
  int n = idx / F2D, f = idx - n * F2D;
  const float* ap = acc + (size_t)n * C2;
  float m = (ap[f] + ap[F2D + f] + ap[2 * F2D + f]) * (1.f / 3.f) + bias[f];
  out[idx] = m > 0.f ? m : 0.f;
}

// ---------------- FFN head: [N,32] -> relu([N,16]) -> [N,2], weights staged in LDS --------
__global__ void k_ffn(const float* __restrict__ x, const float* __restrict__ W1,
                      const float* __restrict__ b1, const float* __restrict__ W2,
                      const float* __restrict__ b2, float* __restrict__ out, int N) {
  __shared__ float sW1[F2D * 16];
  __shared__ float sW2[16 * 2];
  __shared__ float sb1[16];
  __shared__ float sb2[2];
  for (int i = threadIdx.x; i < F2D * 16; i += blockDim.x) sW1[i] = W1[i];
  if (threadIdx.x < 32) sW2[threadIdx.x] = W2[threadIdx.x];
  if (threadIdx.x < 16) sb1[threadIdx.x] = b1[threadIdx.x];
  if (threadIdx.x < 2)  sb2[threadIdx.x] = b2[threadIdx.x];
  __syncthreads();

  int n = blockIdx.x * blockDim.x + threadIdx.x;
  if (n >= N) return;
  const float* xp = x + (size_t)n * F2D;
  float xin[F2D];
  #pragma unroll
  for (int i = 0; i < F2D; ++i) xin[i] = xp[i];

  float hid[16];
  #pragma unroll
  for (int j = 0; j < 16; ++j) {
    float s = sb1[j];
    #pragma unroll
    for (int i = 0; i < F2D; ++i) s += xin[i] * sW1[i * 16 + j];
    hid[j] = s > 0.f ? s : 0.f;
  }
  #pragma unroll
  for (int o = 0; o < 2; ++o) {
    float s = sb2[o];
    #pragma unroll
    for (int j = 0; j < 16; ++j) s += hid[j] * sW2[j * 2 + o];
    out[(size_t)n * 2 + o] = s;
  }
}

// ---------------- host ----------------
extern "C" void kernel_launch(void* const* d_in, const int* in_sizes, int n_in,
                              void* d_out, int out_size, void* d_ws, size_t ws_size,
                              hipStream_t stream) {
  const float* x   = (const float*)d_in[0];
  const int*   ei  = (const int*)  d_in[1];
  const float* W1  = (const float*)d_in[2];
  const float* as1 = (const float*)d_in[3];
  const float* ad1 = (const float*)d_in[4];
  const float* b1  = (const float*)d_in[5];
  const float* W2  = (const float*)d_in[6];
  const float* as2 = (const float*)d_in[7];
  const float* ad2 = (const float*)d_in[8];
  const float* b2  = (const float*)d_in[9];
  const float* fW1 = (const float*)d_in[10];
  const float* fb1 = (const float*)d_in[11];
  const float* fW2 = (const float*)d_in[12];
  const float* fb2 = (const float*)d_in[13];

  const int N    = in_sizes[0] / FIN;
  const int E    = in_sizes[1] / 2;
  const int Etot = E + N;
  const int* src = ei;
  const int* dst = ei + E;

  char* ws = (char*)d_ws;
  size_t off = 0;
  auto alloc = [&](size_t bytes) -> void* {
    off = (off + 255) & ~(size_t)255;
    void* p = ws + off;
    off += bytes;
    return p;
  };

  _Float16* xh   = (_Float16*)alloc((size_t)N * FIN * 2);   // x in f16
  _Float16* g1h  = (_Float16*)alloc((size_t)N * C1 * 2);    // relu(layer1 out) in f16
  _Float16* W1p  = (_Float16*)alloc((size_t)FIN * C1 * 2);  // packed B fragments
  _Float16* W2p  = (_Float16*)alloc((size_t)C1 * C2 * 2);
  float* bufH    = (float*)alloc((size_t)N * C1 * 4);       // h1, later accum2
  float* bufACC  = (float*)alloc((size_t)N * C1 * 4);       // accum1, later h2
  float* a_s     = (float*)alloc((size_t)N * HH * 4);
  float* a_d     = (float*)alloc((size_t)N * HH * 4);
  float* emax    = (float*)alloc((size_t)N * HH * 4);
  float* denom   = (float*)alloc((size_t)N * HH * 4);
  float* elog    = (float*)alloc((size_t)Etot * HH * 4);
  float* out2    = (float*)alloc((size_t)N * F2D * 4);

  const int BT = 256;
  const uint32_t NEG_INF = 0xFF800000u;
  const int gemmBlocks = CDIV(CDIV(N, 16), 4);              // 4 waves / 128-thread block

  // conversions + weight packing
  k_f32_to_f16<<<CDIV(N * FIN, BT), BT, 0, stream>>>(x, xh, N * FIN);
  k_pack_B<<<CDIV(FIN * C1, BT), BT, 0, stream>>>(W1, W1p, FIN, C1);
  k_pack_B<<<CDIV(C1 * C2, BT), BT, 0, stream>>>(W2, W2p, C1, C2);

  // ===== layer 1 =====
  k_gemm_wmma<4><<<gemmBlocks, 128, 0, stream>>>(xh, W1p, bufH, N);
  k_attn_scores<<<CDIV(N * HH, BT), BT, 0, stream>>>(bufH, as1, ad1, a_s, a_d, N, 32);
  k_fill_u32<<<CDIV(N * HH, BT), BT, 0, stream>>>((uint32_t*)emax,  NEG_INF, N * HH);
  k_fill_u32<<<CDIV(N * HH, BT), BT, 0, stream>>>((uint32_t*)denom, 0u,      N * HH);
  k_fill_u32<<<CDIV(N * C1, BT), BT, 0, stream>>>((uint32_t*)bufACC, 0u,     N * C1);
  k_edge_max      <<<CDIV(Etot, BT), BT, 0, stream>>>(src, dst, E, Etot, a_s, a_d, elog, emax);
  k_edge_expsum   <<<CDIV(Etot, BT), BT, 0, stream>>>(src, dst, E, Etot, emax, elog, denom);
  k_edge_aggregate<<<CDIV(Etot * HH, BT), BT, 0, stream>>>(src, dst, E, Etot, elog, denom,
                                                           bufH, bufACC);
  k_finalize_concat<<<CDIV(N * C1, BT), BT, 0, stream>>>(bufACC, b1, g1h, N);

  // ===== layer 2 =====
  k_gemm_wmma<3><<<gemmBlocks, 128, 0, stream>>>(g1h, W2p, bufACC, N);   // h2 -> bufACC
  k_attn_scores<<<CDIV(N * HH, BT), BT, 0, stream>>>(bufACC, as2, ad2, a_s, a_d, N, 32);
  k_fill_u32<<<CDIV(N * HH, BT), BT, 0, stream>>>((uint32_t*)emax,  NEG_INF, N * HH);
  k_fill_u32<<<CDIV(N * HH, BT), BT, 0, stream>>>((uint32_t*)denom, 0u,      N * HH);
  k_fill_u32<<<CDIV(N * C2, BT), BT, 0, stream>>>((uint32_t*)bufH, 0u,      N * C2); // accum2
  k_edge_max      <<<CDIV(Etot, BT), BT, 0, stream>>>(src, dst, E, Etot, a_s, a_d, elog, emax);
  k_edge_expsum   <<<CDIV(Etot, BT), BT, 0, stream>>>(src, dst, E, Etot, emax, elog, denom);
  k_edge_aggregate<<<CDIV(Etot * HH, BT), BT, 0, stream>>>(src, dst, E, Etot, elog, denom,
                                                           bufACC, bufH);
  k_finalize_mean<<<CDIV(N * F2D, BT), BT, 0, stream>>>(bufH, b2, out2, N);

  // ===== FFN head =====
  k_ffn<<<CDIV(N, BT), BT, 0, stream>>>(out2, fW1, fb1, fW2, fb2, (float*)d_out, N);
}
